// Model_57982058496054
// MI455X (gfx1250) — compile-verified, matrix operand order checked by reading, not run on hardware
//
#include <hip/hip_runtime.h>

// ---------------------------------------------------------------------------
// CDNA5 / gfx1250 transformer block: RMSNorm -> QKV -> attention -> Wo+res ->
// RMSNorm -> SwiGLU MLP + res.  Matmuls: v_wmma_f32_16x16x32_bf16.
// Tile staging: global_load_async_to_lds_b128 (ASYNCcnt), double-buffered in
// the GEMM so the async DMA of tile k+1 overlaps WMMA on tile k.
// ---------------------------------------------------------------------------

typedef __attribute__((ext_vector_type(16))) __bf16 v16bf;
typedef __attribute__((ext_vector_type(8)))  __bf16 v8bf;
typedef __attribute__((ext_vector_type(8)))  float  v8f;

#define D_MODEL 4096
#define D_FF    11008
#define SEQ     2048
#define NBATCH  2
#define NTOK    (NBATCH * SEQ)   // 4096
#define NHEADS  32
#define HEADDIM 128

// ---------------- bf16 helpers (explicit RNE, no __bf16 arithmetic) --------
__device__ __forceinline__ __bf16 f2bf(float f) {
    union { float f; unsigned u; } a; a.f = f;
    unsigned r = a.u + 0x7FFFu + ((a.u >> 16) & 1u);
    union { unsigned short s; __bf16 b; } o; o.s = (unsigned short)(r >> 16);
    return o.b;
}
__device__ __forceinline__ float bf2f(__bf16 b) {
    union { __bf16 b; unsigned short s; } i; i.b = b;
    union { unsigned u; float f; } o; o.u = ((unsigned)i.s) << 16;
    return o.f;
}

// Load a 32-byte WMMA fragment (16 bf16) from LDS/global as two b128 loads.
__device__ __forceinline__ v16bf load_frag(const __bf16* p) {
    v8bf lo = *(const v8bf*)(p);
    v8bf hi = *(const v8bf*)(p + 8);
    return __builtin_shufflevector(lo, hi, 0,1,2,3,4,5,6,7,8,9,10,11,12,13,14,15);
}

__device__ __forceinline__ v8f wmma_bf16(v16bf a, v16bf b, v8f c) {
    return __builtin_amdgcn_wmma_f32_16x16x32_bf16(
        /*neg_a=*/false, a, /*neg_b=*/false, b,
        /*c_mod=*/(short)0, c, /*reuse_a=*/false, /*reuse_b=*/false);
}

// ---- CDNA5 async copy: 64B global -> LDS per lane (ASYNCcnt tracked) ------
// INST_OFFSET is added to both LDS and global addresses (ISA 08 §4.4).
// LDS byte offset = low 32 bits of the generic pointer (ISA §10.2 aperture).
__device__ __forceinline__ void async_copy64(void* lds, const void* g) {
    unsigned l = (unsigned)(unsigned long long)lds;
    asm volatile(
        "global_load_async_to_lds_b128 %0, %1, off\n\t"
        "global_load_async_to_lds_b128 %0, %1, off offset:16\n\t"
        "global_load_async_to_lds_b128 %0, %1, off offset:32\n\t"
        "global_load_async_to_lds_b128 %0, %1, off offset:48"
        :: "v"(l), "v"(g) : "memory");
}
__device__ __forceinline__ void wait_async0() {
    asm volatile("s_wait_asynccnt 0x0" ::: "memory");
}

// ---------------- weight convert + transpose: Wt[n][k] = bf16(W[k][n]) -----
__global__ void __launch_bounds__(256) convert_w_kernel(
        const float* __restrict__ W, __bf16* __restrict__ Wt, int K, int N) {
    size_t total = (size_t)K * (size_t)N;
    size_t stride = (size_t)gridDim.x * blockDim.x;
    for (size_t o = (size_t)blockIdx.x * blockDim.x + threadIdx.x; o < total; o += stride) {
        size_t n = o / (size_t)K;
        size_t k = o - n * (size_t)K;
        Wt[o] = f2bf(W[k * (size_t)N + n]);
    }
}

// ---------------- RMSNorm: fp32 row -> bf16 row ----------------------------
__global__ void __launch_bounds__(256) rmsnorm_kernel(
        const float* __restrict__ x, __bf16* __restrict__ out, int D) {
    __shared__ float red[256];
    int row = blockIdx.x;
    const float* px = x + (size_t)row * D;
    float ss = 0.f;
    for (int i = threadIdx.x; i < D; i += 256) { float v = px[i]; ss += v * v; }
    red[threadIdx.x] = ss;
    __syncthreads();
    for (int s = 128; s > 0; s >>= 1) {
        if (threadIdx.x < s) red[threadIdx.x] += red[threadIdx.x + s];
        __syncthreads();
    }
    float scale = __frsqrt_rn(red[0] / (float)D + 1e-6f);
    __bf16* po = out + (size_t)row * D;
    for (int i = threadIdx.x; i < D; i += 256) po[i] = f2bf(px[i] * scale);
}

// ---------------- bf16 WMMA GEMM: C[M][N] = A[M][K] * Bt[N][K]^T -----------
// 128x128 block tile, BK=32, 8 waves, each wave = 32x64 (2x4 WMMA tiles).
// Double-buffered LDS: async DMA of slice k+1 overlaps WMMA on slice k.
#define EPI_BF16   0   // out bf16
#define EPI_SILU   1   // out bf16 = silu(c)
#define EPI_MUL    2   // out bf16 = c * aux_bf16
#define EPI_RESF32 3   // out f32  = c + aux_f32

#define GEMM_STR 40    // LDS row stride in elements (80B, conflict-free)

template <int EPI>
__global__ void __launch_bounds__(256, 1) gemm_bf16_kernel(
        const __bf16* __restrict__ A, const __bf16* __restrict__ Bt,
        void* __restrict__ out, const void* __restrict__ aux, int K, int N) {
    __shared__ __bf16 sA[2][128 * GEMM_STR];
    __shared__ __bf16 sB[2][128 * GEMM_STR];

    const int tid  = threadIdx.x;
    const int lane = tid & 31;
    const int wid  = tid >> 5;
    const int wm   = wid & 3;      // 4 M slots of 32 rows
    const int wn   = wid >> 2;     // 2 N slots of 64 cols
    const int m0   = blockIdx.y * 128;
    const int n0   = blockIdx.x * 128;
    const int hl   = lane >> 4;    // half-wave selector
    const int ml   = lane & 15;

    // loader role: threads 0-127 -> A rows, 128-255 -> B rows (64B per row)
    const bool loadA = tid < 128;
    const int lrow = loadA ? tid : (tid - 128);
    const __bf16* gsrc = loadA ? (A  + (size_t)(m0 + lrow) * K)
                               : (Bt + (size_t)(n0 + lrow) * K);
    __bf16* ldst[2] = { (loadA ? sA[0] : sB[0]) + lrow * GEMM_STR,
                        (loadA ? sA[1] : sB[1]) + lrow * GEMM_STR };

    v8f acc[2][4];
    const v8f vzero = {0.f,0.f,0.f,0.f,0.f,0.f,0.f,0.f};
    for (int i = 0; i < 2; ++i)
        for (int j = 0; j < 4; ++j) acc[i][j] = vzero;

    const int ksteps = K >> 5;

    // prologue: fill buffer 0 with slice 0
    async_copy64(ldst[0], gsrc);
    wait_async0();
    __syncthreads();

    int p = 0;
    for (int kt = 0; kt < ksteps; ++kt) {
        // kick off DMA of the next K slice into the other buffer
        if (kt + 1 < ksteps) {
            async_copy64(ldst[p ^ 1], gsrc + (size_t)(kt + 1) * 32);
            if (kt + 4 < ksteps)   // warm GL2 a few slices ahead
                __builtin_prefetch(gsrc + (size_t)(kt + 4) * 32, 0, 1);
        }

        // compute on the current buffer while the DMA flies
        const __bf16* cA = sA[p];
        const __bf16* cB = sB[p];
        v16bf af[2], bfg[4];
        for (int i = 0; i < 2; ++i)
            af[i]  = load_frag(cA + (wm * 32 + i * 16 + ml) * GEMM_STR + hl * 16);
        for (int j = 0; j < 4; ++j)
            bfg[j] = load_frag(cB + (wn * 64 + j * 16 + ml) * GEMM_STR + hl * 16);
        for (int i = 0; i < 2; ++i)
            for (int j = 0; j < 4; ++j)
                acc[i][j] = wmma_bf16(af[i], bfg[j], acc[i][j]);

        wait_async0();        // my async writes of buffer p^1 landed
        __syncthreads();      // everyone done reading buffer p
        p ^= 1;
    }

    // epilogue: C layout -> lanes 0-15 hold row r, lanes 16-31 hold row 8+r
    for (int i = 0; i < 2; ++i) {
        for (int j = 0; j < 4; ++j) {
            int row0 = m0 + wm * 32 + i * 16 + hl * 8;
            int col  = n0 + wn * 64 + j * 16 + ml;
            for (int r = 0; r < 8; ++r) {
                float c = acc[i][j][r];
                size_t idx = (size_t)(row0 + r) * N + col;
                if (EPI == EPI_BF16) {
                    ((__bf16*)out)[idx] = f2bf(c);
                } else if (EPI == EPI_SILU) {
                    float s = c / (1.f + __expf(-c));
                    ((__bf16*)out)[idx] = f2bf(s);
                } else if (EPI == EPI_MUL) {
                    float g = bf2f(((const __bf16*)aux)[idx]);
                    ((__bf16*)out)[idx] = f2bf(c * g);
                } else { // EPI_RESF32
                    ((float*)out)[idx] = c + ((const float*)aux)[idx];
                }
            }
        }
    }
}

// ---------------- flash attention (bf16 WMMA, fp32 online softmax) ---------
// grid = (SEQ/128, NBATCH*NHEADS), block = 256 (8 waves x 16 queries)
#define KSTR 136   // K tile LDS row stride (272B -> conflict-free)
#define VSTR 72    // V^T / P LDS row stride (144B -> conflict-free)

__global__ void __launch_bounds__(256, 1) attn_kernel(
        const __bf16* __restrict__ Q, const __bf16* __restrict__ Kb,
        const __bf16* __restrict__ Vb, __bf16* __restrict__ O) {
    __shared__ __bf16 sK[64 * KSTR];        // K tile  [kv=64][hd=128]
    __shared__ __bf16 sVT[128 * VSTR];      // V tile^T [hd=128][kv=64]
    __shared__ __bf16 sP[8][16 * VSTR];     // per-wave P tile [16][64]

    const int tid  = threadIdx.x;
    const int lane = tid & 31;
    const int wid  = tid >> 5;
    const int ml   = lane & 15;
    const int hl   = lane >> 4;

    const int bh = blockIdx.y;
    const int b  = bh >> 5;
    const int h  = bh & 31;
    const size_t base = ((size_t)b * SEQ) * D_MODEL + (size_t)h * HEADDIM;
    const int q0 = blockIdx.x * 128 + wid * 16;
    const float scale = 0.08838834764831845f;   // 1/sqrt(128)

    // preload Q fragments straight from global (lane ml -> query row q0+ml)
    v16bf qf[4];
    const __bf16* qrow = Q + base + (size_t)(q0 + ml) * D_MODEL;
    for (int kc = 0; kc < 4; ++kc)
        qf[kc] = load_frag(qrow + kc * 32 + hl * 16);

    const v8f vzero = {0.f,0.f,0.f,0.f,0.f,0.f,0.f,0.f};
    v8f oacc[8];
    for (int j = 0; j < 8; ++j) oacc[j] = vzero;
    float mrow[8], srow[8];
    for (int r = 0; r < 8; ++r) { mrow[r] = -1e30f; srow[r] = 0.f; }

    for (int kv0 = 0; kv0 < SEQ; kv0 += 64) {
        __syncthreads();
        { // stage K tile via async-to-LDS: thread -> (row, 32-element quarter)
            int r = tid >> 2, c = (tid & 3) * 32;
            async_copy64(sK + r * KSTR + c,
                         Kb + base + (size_t)(kv0 + r) * D_MODEL + c);
        }
        { // stage V transposed (element scatter, overlaps the async K DMA)
            int r = tid >> 2, c0 = (tid & 3) * 32;
            const __bf16* s = Vb + base + (size_t)(kv0 + r) * D_MODEL + c0;
            union { v8bf v[4]; __bf16 e[32]; } buf;
            const v8bf* s8 = (const v8bf*)s;
            buf.v[0] = s8[0]; buf.v[1] = s8[1]; buf.v[2] = s8[2]; buf.v[3] = s8[3];
            for (int i = 0; i < 32; ++i) sVT[(c0 + i) * VSTR + r] = buf.e[i];
        }
        wait_async0();
        __syncthreads();

        // scores: S[16 x 64] = Q(16x128) * K^T
        v8f sacc[4];
        for (int nt = 0; nt < 4; ++nt) sacc[nt] = vzero;
        for (int nt = 0; nt < 4; ++nt)
            for (int kc = 0; kc < 4; ++kc) {
                v16bf bfrag = load_frag(sK + (nt * 16 + ml) * KSTR + kc * 32 + hl * 16);
                sacc[nt] = wmma_bf16(qf[kc], bfrag, sacc[nt]);
            }

        // online softmax; row r (or 8+r) lives in lanes of a 16-lane group
        for (int r = 0; r < 8; ++r) {
            float mx = mrow[r];
            for (int nt = 0; nt < 4; ++nt) mx = fmaxf(mx, sacc[nt][r] * scale);
            for (int off = 8; off >= 1; off >>= 1)
                mx = fmaxf(mx, __shfl_xor(mx, off, 32));
            float alpha = __expf(mrow[r] - mx);
            mrow[r] = mx;
            srow[r] *= alpha;
            for (int j = 0; j < 8; ++j) oacc[j][r] *= alpha;
            float psum = 0.f;
            for (int nt = 0; nt < 4; ++nt) {
                float p = __expf(sacc[nt][r] * scale - mx);
                psum += p;
                sP[wid][(hl * 8 + r) * VSTR + nt * 16 + ml] = f2bf(p);
            }
            for (int off = 8; off >= 1; off >>= 1)
                psum += __shfl_xor(psum, off, 32);
            srow[r] += psum;
        }

        // O += P(16x64) * V(64x128); same-wave LDS ops are in-order (ISA 7.3)
        for (int kc = 0; kc < 2; ++kc) {
            v16bf pf = load_frag(&sP[wid][ml * VSTR + kc * 32 + hl * 16]);
            for (int j = 0; j < 8; ++j) {
                v16bf vfrag = load_frag(sVT + (j * 16 + ml) * VSTR + kc * 32 + hl * 16);
                oacc[j] = wmma_bf16(pf, vfrag, oacc[j]);
            }
        }
    }

    // normalize and write attn output (bf16, token-major [tok][h*128+d])
    for (int j = 0; j < 8; ++j)
        for (int r = 0; r < 8; ++r) {
            float o = oacc[j][r] / srow[r];
            O[base + (size_t)(q0 + hl * 8 + r) * D_MODEL + j * 16 + ml] = f2bf(o);
        }
}

// ---------------------------------------------------------------------------
extern "C" void kernel_launch(void* const* d_in, const int* in_sizes, int n_in,
                              void* d_out, int out_size, void* d_ws, size_t ws_size,
                              hipStream_t stream) {
    const float* x  = (const float*)d_in[0];
    const float* wq = (const float*)d_in[1];
    const float* wk = (const float*)d_in[2];
    const float* wv = (const float*)d_in[3];
    const float* wo = (const float*)d_in[4];
    const float* wg = (const float*)d_in[5];
    const float* wu = (const float*)d_in[6];
    const float* wd = (const float*)d_in[7];

    // workspace carve-out (256B aligned)
    size_t off = 0;
    auto alloc = [&](size_t bytes) -> void* {
        void* p = (char*)d_ws + off;
        off += (bytes + 255) & ~(size_t)255;
        return p;
    };
    const size_t DD = (size_t)D_MODEL * D_MODEL;
    const size_t DF = (size_t)D_MODEL * D_FF;
    const size_t TD = (size_t)NTOK * D_MODEL;
    const size_t TF = (size_t)NTOK * D_FF;

    __bf16* wq_t = (__bf16*)alloc(DD * 2);
    __bf16* wk_t = (__bf16*)alloc(DD * 2);
    __bf16* wv_t = (__bf16*)alloc(DD * 2);
    __bf16* wo_t = (__bf16*)alloc(DD * 2);
    __bf16* wg_t = (__bf16*)alloc(DF * 2);
    __bf16* wu_t = (__bf16*)alloc(DF * 2);
    __bf16* wd_t = (__bf16*)alloc(DF * 2);
    __bf16* h_bf    = (__bf16*)alloc(TD * 2);
    __bf16* q_bf    = (__bf16*)alloc(TD * 2);
    __bf16* k_bf    = (__bf16*)alloc(TD * 2);
    __bf16* v_bf    = (__bf16*)alloc(TD * 2);
    __bf16* attn_bf = (__bf16*)alloc(TD * 2);
    float*  x1_f    = (float*) alloc(TD * 4);
    __bf16* h2_bf   = (__bf16*)alloc(TD * 2);
    __bf16* g_bf    = (__bf16*)alloc(TF * 2);
    __bf16* p_bf    = (__bf16*)alloc(TF * 2);

    dim3 blk(256);

    // 1. weights -> bf16 transposed [N][K]
    convert_w_kernel<<<2048, blk, 0, stream>>>(wq, wq_t, D_MODEL, D_MODEL);
    convert_w_kernel<<<2048, blk, 0, stream>>>(wk, wk_t, D_MODEL, D_MODEL);
    convert_w_kernel<<<2048, blk, 0, stream>>>(wv, wv_t, D_MODEL, D_MODEL);
    convert_w_kernel<<<2048, blk, 0, stream>>>(wo, wo_t, D_MODEL, D_MODEL);
    convert_w_kernel<<<4096, blk, 0, stream>>>(wg, wg_t, D_MODEL, D_FF);
    convert_w_kernel<<<4096, blk, 0, stream>>>(wu, wu_t, D_MODEL, D_FF);
    convert_w_kernel<<<4096, blk, 0, stream>>>(wd, wd_t, D_FF, D_MODEL);

    // 2. h = rmsnorm(x)
    rmsnorm_kernel<<<NTOK, blk, 0, stream>>>(x, h_bf, D_MODEL);

    // 3. q/k/v projections
    dim3 gdd(D_MODEL / 128, NTOK / 128);
    gemm_bf16_kernel<EPI_BF16><<<gdd, blk, 0, stream>>>(h_bf, wq_t, q_bf, nullptr, D_MODEL, D_MODEL);
    gemm_bf16_kernel<EPI_BF16><<<gdd, blk, 0, stream>>>(h_bf, wk_t, k_bf, nullptr, D_MODEL, D_MODEL);
    gemm_bf16_kernel<EPI_BF16><<<gdd, blk, 0, stream>>>(h_bf, wv_t, v_bf, nullptr, D_MODEL, D_MODEL);

    // 4. attention
    attn_kernel<<<dim3(SEQ / 128, NBATCH * NHEADS), blk, 0, stream>>>(q_bf, k_bf, v_bf, attn_bf);

    // 5. x1 = x + attn @ wo
    gemm_bf16_kernel<EPI_RESF32><<<gdd, blk, 0, stream>>>(attn_bf, wo_t, x1_f, x, D_MODEL, D_MODEL);

    // 6. h2 = rmsnorm(x1)
    rmsnorm_kernel<<<NTOK, blk, 0, stream>>>(x1_f, h2_bf, D_MODEL);

    // 7/8. SwiGLU: g = silu(h2@wg); p = (h2@wu) * g
    dim3 gdf(D_FF / 128, NTOK / 128);
    gemm_bf16_kernel<EPI_SILU><<<gdf, blk, 0, stream>>>(h2_bf, wg_t, g_bf, nullptr, D_MODEL, D_FF);
    gemm_bf16_kernel<EPI_MUL> <<<gdf, blk, 0, stream>>>(h2_bf, wu_t, p_bf, g_bf, D_MODEL, D_FF);

    // 9. out = x1 + p @ wd
    gemm_bf16_kernel<EPI_RESF32><<<gdd, blk, 0, stream>>>(p_bf, wd_t, (float*)d_out, x1_f, D_FF, D_MODEL);
}
